// InceptionScore_45183055954015
// MI455X (gfx1250) — matrix operand order repaired and checked by reading, not compiled, not run on hardware
//
#include <hip/hip_runtime.h>
#include <hip/hip_bf16.h>
#include <stdint.h>

// ---------------------------------------------------------------------------
// Inception-Score on MI455X (gfx1250).
//
// KL reduction:  mean_r KL(p_r || m) = mean_r(NE_r) - sum_c m[c] log m[c]
// NE_r = sum_c p log p is permutation-invariant (computed once).  Split means
// m = S x P (S one-hot) are computed with v_wmma_f32_16x16x32_bf16.
// P kept as bf16 (~100MB, L2-resident).  Four repetitions are batched per
// wave so each B tile load is amortized over 4 WMMAs -> L2 traffic ~2.5 GB.
//
// Workspace requirement: ~113 MB (layout in kernel_launch).
// ---------------------------------------------------------------------------

#define N_ROWS       50000
#define N_CLASSES    1000
#define ROWS_P       50016      // K padded to multiple of 32 (50016 = 32*1563)
#define CLASSES_P    1008       // N padded to multiple of 16 (63 tiles)
#define N_REPS       100
#define N_SPLITS     10
#define STEP         5000       // rows per split
#define M_SPLITS_P   16         // M padded to 16 for WMMA
#define K_ITERS      (ROWS_P / 32)   // 1563
#define REPS_PER_CTA 4          // reps batched per wave (B-tile reuse)

typedef __attribute__((ext_vector_type(16))) __bf16 v16bf;
typedef __attribute__((ext_vector_type(8)))  float  v8f;

// ---------------------------------------------------------------------------
// small helpers
// ---------------------------------------------------------------------------
__device__ __forceinline__ uint32_t hash32(uint32_t x) {
    x ^= x >> 16; x *= 0x7feb352du;
    x ^= x >> 15; x *= 0x846ca68bu;
    x ^= x >> 16;
    return x;
}

// 4-round Feistel bijection on 16-bit domain, keyed per repetition.
__device__ __forceinline__ uint32_t feistel16(uint32_t x,
                                              uint32_t k0, uint32_t k1,
                                              uint32_t k2, uint32_t k3) {
    uint32_t L = x & 0xFFu, R = (x >> 8) & 0xFFu, t;
    t = R; R = L ^ (hash32(R ^ k0) & 0xFFu); L = t;
    t = R; R = L ^ (hash32(R ^ k1) & 0xFFu); L = t;
    t = R; R = L ^ (hash32(R ^ k2) & 0xFFu); L = t;
    t = R; R = L ^ (hash32(R ^ k3) & 0xFFu); L = t;
    return (R << 8) | L;
}

__device__ __forceinline__ float waveMax(float v) {
    #pragma unroll
    for (int o = 16; o > 0; o >>= 1) v = fmaxf(v, __shfl_xor(v, o, 32));
    return v;
}
__device__ __forceinline__ float waveSum(float v) {
    #pragma unroll
    for (int o = 16; o > 0; o >>= 1) v += __shfl_xor(v, o, 32);
    return v;
}

// ---------------------------------------------------------------------------
// Kernel 1: softmax per row; writes P^T (bf16, [CLASSES_P][ROWS_P]) and
// per-row negentropy NE_r = sum p*log p.  One wave (32 lanes) per row.
// ---------------------------------------------------------------------------
__global__ __launch_bounds__(256)
void softmax_rows_kernel(const float* __restrict__ logits,
                         __hip_bfloat16* __restrict__ pT,
                         float* __restrict__ NE) {
    const int lane = threadIdx.x & 31;
    const int wave = threadIdx.x >> 5;
    const int row  = blockIdx.x * 8 + wave;
    if (row >= N_ROWS) return;

    const float* __restrict__ x = logits + (size_t)row * N_CLASSES;

    float vals[32];
    float mx = -3.4e38f;
    #pragma unroll
    for (int j = 0; j < 32; ++j) {
        int c = lane + j * 32;
        float v = (c < N_CLASSES) ? x[c] : -3.4e38f;
        vals[j] = v;
        mx = fmaxf(mx, v);
    }
    mx = waveMax(mx);

    float s = 0.f;
    #pragma unroll
    for (int j = 0; j < 32; ++j) {
        int c = lane + j * 32;
        float e = (c < N_CLASSES) ? expf(vals[j] - mx) : 0.f;
        vals[j] = e;
        s += e;
    }
    s = waveSum(s);
    const float inv = 1.f / s;

    float ne = 0.f;
    #pragma unroll
    for (int j = 0; j < 32; ++j) {
        int c = lane + j * 32;
        if (c < N_CLASSES) {
            float p = vals[j] * inv;
            if (p > 0.f) ne += p * logf(p);
            pT[(size_t)c * ROWS_P + row] = __float2bfloat16(p);
        }
    }
    ne = waveSum(ne);
    if (lane == 0) NE[row] = ne;
}

// ---------------------------------------------------------------------------
// Kernel 2: per-rep balanced split assignment via Feistel cycle-walking
// permutation of [0, N_ROWS); also accumulates per-(rep,split) NE sums.
// ---------------------------------------------------------------------------
__global__ __launch_bounds__(256)
void assign_kernel(const float* __restrict__ NE,
                   uint8_t* __restrict__ assign,
                   float* __restrict__ ne_sum) {
    const int r   = blockIdx.x * 256 + threadIdx.x;
    const int rep = blockIdx.y;
    if (r >= ROWS_P) return;

    uint8_t s = 255;  // padding marker (matches no split 0..15)
    if (r < N_ROWS) {
        const uint32_t k0 = hash32(rep * 4u + 0x9E3779B9u);
        const uint32_t k1 = hash32(rep * 4u + 1u + 0x85EBCA6Bu);
        const uint32_t k2 = hash32(rep * 4u + 2u + 0xC2B2AE35u);
        const uint32_t k3 = hash32(rep * 4u + 3u + 0x27D4EB2Fu);
        uint32_t xv = (uint32_t)r;
        do { xv = feistel16(xv, k0, k1, k2, k3); } while (xv >= N_ROWS);
        s = (uint8_t)(xv / STEP);
        atomicAdd(&ne_sum[rep * M_SPLITS_P + s], NE[r]);
    }
    assign[(size_t)rep * ROWS_P + r] = s;
}

// ---------------------------------------------------------------------------
// Kernel 3: split sums m[rep][s][c] = sum_{r in split s} P[r][c] as a
// one-hot x P GEMM using v_wmma_f32_16x16x32_bf16, with REPS_PER_CTA reps
// batched per wave so each B tile is reused by 4 WMMAs.
//   grid.x = class tile (CLASSES_P/16 = 63), grid.y = rep group (25).
//   8 waves per block split the K range, LDS reduction, plain stores.
// ---------------------------------------------------------------------------
__global__ __launch_bounds__(256)
void splitmean_wmma_kernel(const __bf16* __restrict__ pT,
                           const uint8_t* __restrict__ assign,
                           float* __restrict__ mOut) {
    const int lane    = threadIdx.x & 31;
    const int wave    = threadIdx.x >> 5;
    const int repBase = blockIdx.y * REPS_PER_CTA;
    const int cbase   = blockIdx.x * 16;

    const int  n      = lane & 15;          // B/D column within tile
    const bool hiHalf = (lane >= 16);
    const int  kofsB  = hiHalf ? 16 : 0;    // B: K rows held by this lane
    const int  kloA   = hiHalf ? 16 : 0;    // A: first K group
    const int  khiA   = hiHalf ? 24 : 8;    // A: second K group
    const uint32_t mrow = (uint32_t)(lane & 15);  // A row (split id) this lane holds

    const uint8_t* __restrict__ asg0 = assign + (size_t)repBase * ROWS_P;
    const __bf16*  __restrict__ bcol = pT + (size_t)(cbase + n) * ROWS_P;

    v8f acc[REPS_PER_CTA];
    #pragma unroll
    for (int rr = 0; rr < REPS_PER_CTA; ++rr)
        acc[rr] = (v8f){0.f, 0.f, 0.f, 0.f, 0.f, 0.f, 0.f, 0.f};

    union ABf { v16bf v; unsigned short u[16]; };

    for (int it = wave; it < K_ITERS; it += 8) {
        const int kbase = it * 32;

        // B tile: 16 consecutive K values of this lane's class column (32B),
        // shared by all REPS_PER_CTA WMMAs below.
        const __bf16* bptr = bcol + kbase + kofsB;
        v16bf b = *(const v16bf*)bptr;
        // Prefetch this wave's next B tile (8 iterations ahead = +256 rows).
        __builtin_prefetch((const void*)(bptr + 256), 0, 1);

        #pragma unroll
        for (int rr = 0; rr < REPS_PER_CTA; ++rr) {
            const uint8_t* asg = asg0 + (size_t)rr * ROWS_P;
            uint2 q0 = *(const uint2*)(asg + kbase + kloA);
            uint2 q1 = *(const uint2*)(asg + kbase + khiA);
            ABf a;
            #pragma unroll
            for (int i = 0; i < 4; ++i) {
                a.u[i]      = (((q0.x >> (8*i)) & 0xFFu) == mrow) ? 0x3F80 : 0;
                a.u[4 + i]  = (((q0.y >> (8*i)) & 0xFFu) == mrow) ? 0x3F80 : 0;
                a.u[8 + i]  = (((q1.x >> (8*i)) & 0xFFu) == mrow) ? 0x3F80 : 0;
                a.u[12 + i] = (((q1.y >> (8*i)) & 0xFFu) == mrow) ? 0x3F80 : 0;
            }
            acc[rr] = __builtin_amdgcn_wmma_f32_16x16x32_bf16(
                          false, a.v, false, b, (short)0, acc[rr], false, false);
        }
    }

    // Reduce the 8 per-wave 16x16 accumulators (per rep) through LDS.
    __shared__ float red[8][REPS_PER_CTA][256];   // 32 KB
    #pragma unroll
    for (int rr = 0; rr < REPS_PER_CTA; ++rr)
        #pragma unroll
        for (int j = 0; j < 8; ++j)
            red[wave][rr][lane * 8 + j] = acc[rr][j];
    __syncthreads();

    const int t = threadIdx.x;
    const int lane2 = t >> 3, j2 = t & 7;
    const int M = j2 + ((lane2 >= 16) ? 8 : 0);   // split index (0..15)
    const int N = lane2 & 15;                     // class within tile

    #pragma unroll
    for (int rr = 0; rr < REPS_PER_CTA; ++rr) {
        float s = 0.f;
        #pragma unroll
        for (int w = 0; w < 8; ++w) s += red[w][rr][t];
        mOut[(size_t)(repBase + rr) * M_SPLITS_P * CLASSES_P +
             (size_t)M * CLASSES_P + cbase + N] = s;
    }
}

// ---------------------------------------------------------------------------
// Kernel 4: score[rep*10+s] = exp( ne_sum/5000 - sum_c mbar*log(mbar) ),
// mbar = m/5000.  One block per (rep, split).
// ---------------------------------------------------------------------------
__global__ __launch_bounds__(256)
void scores_kernel(const float* __restrict__ m,
                   const float* __restrict__ ne_sum,
                   float* __restrict__ scores) {
    const int b = blockIdx.x;
    const int rep = b / N_SPLITS;
    const int s   = b % N_SPLITS;
    const float* mr = m + (size_t)rep * M_SPLITS_P * CLASSES_P + (size_t)s * CLASSES_P;

    const float invStep = 1.0f / (float)STEP;
    float acc = 0.f;
    for (int c = threadIdx.x; c < N_CLASSES; c += 256) {
        float mm = mr[c] * invStep;
        if (mm > 0.f) acc += mm * logf(mm);
    }
    __shared__ float sd[256];
    sd[threadIdx.x] = acc;
    __syncthreads();
    for (int o = 128; o > 0; o >>= 1) {
        if (threadIdx.x < o) sd[threadIdx.x] += sd[threadIdx.x + o];
        __syncthreads();
    }
    if (threadIdx.x == 0) {
        float kl = ne_sum[rep * M_SPLITS_P + s] * invStep - sd[0];
        scores[b] = expf(kl);
    }
}

// ---------------------------------------------------------------------------
// Kernel 5: mean / population std of the 1000 scores.
// ---------------------------------------------------------------------------
__global__ __launch_bounds__(256)
void stats_kernel(const float* __restrict__ scores, float* __restrict__ out) {
    const int NS = N_REPS * N_SPLITS;
    float s1 = 0.f, s2 = 0.f;
    for (int i = threadIdx.x; i < NS; i += 256) {
        float v = scores[i];
        s1 += v; s2 += v * v;
    }
    __shared__ float a1[256], a2[256];
    a1[threadIdx.x] = s1; a2[threadIdx.x] = s2;
    __syncthreads();
    for (int o = 128; o > 0; o >>= 1) {
        if (threadIdx.x < o) {
            a1[threadIdx.x] += a1[threadIdx.x + o];
            a2[threadIdx.x] += a2[threadIdx.x + o];
        }
        __syncthreads();
    }
    if (threadIdx.x == 0) {
        float mean = a1[0] / (float)NS;
        float var  = a2[0] / (float)NS - mean * mean;
        out[0] = mean;
        out[1] = sqrtf(fmaxf(var, 0.f));
    }
}

// ---------------------------------------------------------------------------
// Host launcher.  Inputs: d_in[0] = logits (f32, 50000*1000);
// d_in[1]=splits(10), d_in[2]=repetitions(100) are compile-time constants of
// the reference and hardcoded here (they are device scalars, unreadable
// during graph capture).  Output: [mean, std] f32.
// ---------------------------------------------------------------------------
static inline size_t alignUp(size_t x) { return (x + 255u) & ~(size_t)255u; }

extern "C" void kernel_launch(void* const* d_in, const int* in_sizes, int n_in,
                              void* d_out, int out_size, void* d_ws, size_t ws_size,
                              hipStream_t stream) {
    (void)in_sizes; (void)n_in; (void)out_size; (void)ws_size;
    const float* logits = (const float*)d_in[0];
    char* ws = (char*)d_ws;

    // Workspace layout (~113 MB total):
    size_t off = 0;
    const size_t szPT  = (size_t)CLASSES_P * ROWS_P * sizeof(__hip_bfloat16); // 100.8 MB
    const size_t offPT = off;                 off = alignUp(off + szPT);
    const size_t szNE  = (size_t)ROWS_P * sizeof(float);                      // 200 KB
    const size_t offNE = off;                 off = alignUp(off + szNE);
    const size_t szASG = (size_t)N_REPS * ROWS_P;                             // 5.0 MB
    const size_t offASG = off;                off = alignUp(off + szASG);
    const size_t szM   = (size_t)N_REPS * M_SPLITS_P * CLASSES_P * sizeof(float); // 6.5 MB
    const size_t offM  = off;                 off = alignUp(off + szM);
    const size_t szNES = (size_t)N_REPS * M_SPLITS_P * sizeof(float);
    const size_t offNES = off;                off = alignUp(off + szNES);
    const size_t szSC  = (size_t)N_REPS * N_SPLITS * sizeof(float);
    const size_t offSC = off;                 off = alignUp(off + szSC);

    __hip_bfloat16* pT   = (__hip_bfloat16*)(ws + offPT);
    float*          NE   = (float*)(ws + offNE);
    uint8_t*        asg  = (uint8_t*)(ws + offASG);
    float*          mBuf = (float*)(ws + offM);
    float*          neS  = (float*)(ws + offNES);
    float*          sc   = (float*)(ws + offSC);

    // Zero the padded P^T (covers pad rows/cols) and the atomic NE sums.
    hipMemsetAsync(pT, 0, szPT, stream);
    hipMemsetAsync(neS, 0, szNES, stream);

    // 1) softmax -> P^T (bf16) + per-row negentropy
    softmax_rows_kernel<<<N_ROWS / 8, 256, 0, stream>>>(logits, pT, NE);

    // 2) per-rep split assignment + NE split sums
    dim3 ga((ROWS_P + 255) / 256, N_REPS);
    assign_kernel<<<ga, 256, 0, stream>>>(NE, asg, neS);

    // 3) split sums via one-hot WMMA GEMM; 4 reps per wave reuse each B tile
    dim3 gg(CLASSES_P / 16, N_REPS / REPS_PER_CTA);
    splitmean_wmma_kernel<<<gg, 256, 0, stream>>>((const __bf16*)pT, asg, mBuf);

    // 4) per-(rep,split) scores
    scores_kernel<<<N_REPS * N_SPLITS, 256, 0, stream>>>(mBuf, neS, sc);

    // 5) mean / population std
    stats_kernel<<<1, 256, 0, stream>>>(sc, (float*)d_out);
}
